// RelationalGCNLayer_9328668966984
// MI455X (gfx1250) — matrix operand order repaired and checked by reading, not compiled, not run on hardware
//
#include <hip/hip_runtime.h>
#include <hip/hip_bf16.h>

// ---------------------------------------------------------------------------
// RelationalGCNLayer for MI455X (gfx1250, wave32, WMMA bf16 + async-to-LDS)
// ---------------------------------------------------------------------------
#define NN      100000
#define DD      256
#define RR      8
#define EE      40000
#define BB      4
#define LN_EPS  1e-5f

// padded LDS row stride for a 256-bf16 row: 528 bytes (132 dwords) ->
// 16 consecutive rows land on distinct bank groups (stride 4 banks).
#define ROWB    528
#define SMEM_B  (256 * ROWB)              // 135168: Wt tile (K=256 x N=256 bf16)
#define SMEM_A  (16 * ROWB)               // 8448:   one A tile (M=16 x K=256 bf16)
#define SMEM_SELF (SMEM_B + 2 * SMEM_A)                   // 152064 (double-buffered A)
#define SMEM_EDGE (SMEM_B + 2 * SMEM_A + 2 * 16 * 8)      // 152320 (+ meta[2])

typedef __attribute__((ext_vector_type(16))) __bf16 v16bf;
typedef __attribute__((ext_vector_type(8)))  float  v8f;

union Frag { v16bf v; uint4 q[2]; };

__device__ __forceinline__ unsigned int pk2(float a, float b) {
    unsigned int ua = __float_as_uint(a); ua += 0x7fffu + ((ua >> 16) & 1u);
    unsigned int ub = __float_as_uint(b); ub += 0x7fffu + ((ub >> 16) & 1u);
    return (ua >> 16) | (ub & 0xffff0000u);
}
__device__ __forceinline__ unsigned short bf16of(float a) {
    unsigned int ua = __float_as_uint(a); ua += 0x7fffu + ((ua >> 16) & 1u);
    return (unsigned short)(ua >> 16);
}

// Async copy of one 16B chunk: global -> LDS (per-lane addresses, ASYNCcnt).
__device__ __forceinline__ void async_b128(unsigned ldsOff, unsigned long long gaddr) {
    asm volatile("global_load_async_to_lds_b128 %0, %1, off"
                 :: "v"(ldsOff), "v"(gaddr) : "memory");
}
__device__ __forceinline__ void wait_async_le2() {
    asm volatile("s_wait_asynccnt 0x2" ::: "memory");
}
__device__ __forceinline__ void wait_async_0() {
    asm volatile("s_wait_asynccnt 0x0" ::: "memory");
}

// ---------------------------------------------------------------------------
// x (f32) -> xb (bf16), row-major N x 256 (512B rows), done once.
// ---------------------------------------------------------------------------
__global__ __launch_bounds__(256) void xcvt_kernel(
    const float* __restrict__ x, unsigned short* __restrict__ xb) {
    int i = blockIdx.x * 256 + threadIdx.x;          // 8 elements per thread
    if (i < NN * DD / 8) {
        const float4* p = (const float4*)(x + (size_t)i * 8);
        float4 a = p[0], b = p[1];
        *(uint4*)(xb + (size_t)i * 8) =
            make_uint4(pk2(a.x, a.y), pk2(a.z, a.w), pk2(b.x, b.y), pk2(b.z, b.w));
    }
}

// ---------------------------------------------------------------------------
// prep: Wt[r][o][d] = (sum_b coef[r,b]*bases[b,d,o]) as bf16 ;  WselfT[o][d]
// ---------------------------------------------------------------------------
__global__ __launch_bounds__(256) void prep_kernel(
    const float* __restrict__ bases, const float* __restrict__ coef,
    const float* __restrict__ wself,
    unsigned short* __restrict__ wt, unsigned short* __restrict__ wst) {
    int gid = blockIdx.x * 256 + threadIdx.x;
    if (gid < RR * 65536) {
        int r = gid >> 16, rem = gid & 65535;
        int d = rem >> 8, o = rem & 255;
        float acc = 0.f;
#pragma unroll
        for (int b = 0; b < BB; ++b)
            acc += coef[r * BB + b] * bases[(b << 16) + (d << 8) + o];
        wt[(r << 16) + (o << 8) + d] = bf16of(acc);
    } else {
        int rem = gid - RR * 65536;
        if (rem < 65536) {
            int d = rem >> 8, o = rem & 255;
            wst[(o << 8) + d] = bf16of(wself[rem]);
        }
    }
}

__global__ void zero_kernel(float* __restrict__ p, int n) {
    for (int i = blockIdx.x * blockDim.x + threadIdx.x; i < n;
         i += gridDim.x * blockDim.x) p[i] = 0.f;
}

__global__ __launch_bounds__(256) void deg_kernel(
    const long long* __restrict__ eidx, float* __restrict__ deg) {
    int e = blockIdx.x * 256 + threadIdx.x;
    int r = blockIdx.y;
    if (e < EE) {
        int d = (int)eidx[(size_t)r * 2 * EE + EE + e];
        atomicAdd(deg + r * NN + d, 1.0f);
    }
}

// issue one A tile (16 rows x 512B) via async-to-LDS: wave w stages rows 2w,2w+1
// row0/row1 are the GLOBAL bf16 row indices (node ids), lane covers 16B chunks.
__device__ __forceinline__ void issue_a_tile(
    const char* xb, char* bufBase, int wave, int lane, int row0g, int row1g) {
    unsigned lds0 = (unsigned)(size_t)(bufBase + (2 * wave) * ROWB + lane * 16);
    unsigned lds1 = lds0 + ROWB;
    unsigned long long g0 =
        (unsigned long long)(size_t)(xb + (size_t)row0g * 512 + lane * 16);
    unsigned long long g1 =
        (unsigned long long)(size_t)(xb + (size_t)row1g * 512 + lane * 16);
    async_b128(lds0, g0);
    async_b128(lds1, g1);
}

// ---------------------------------------------------------------------------
// self GEMM: out[m, :] = bf16(x[m, :]) @ bf16(w_self) + b_self   (f32 accum)
// ---------------------------------------------------------------------------
__global__ __launch_bounds__(256) void self_gemm_kernel(
    const unsigned short* __restrict__ xb, const unsigned short* __restrict__ wst,
    const float* __restrict__ bself, float* __restrict__ out) {
    extern __shared__ char smem[];
    char* smemB  = smem;
    char* smemA0 = smem + SMEM_B;      // buffer p at smemA0 + p*SMEM_A

    // stage w_selfT into LDS once (8B chunks, padded rows)
    const unsigned long long* wg = (const unsigned long long*)wst;
    for (int c = threadIdx.x; c < 16384; c += 256)
        *(unsigned long long*)(smemB + (c >> 6) * ROWB + (c & 63) * 8) = wg[c];

    const int lane = threadIdx.x & 31;
    const int wave = threadIdx.x >> 5;
    const int mrow = lane & 15;
    const int half = lane >> 4;
    const int n0 = wave * 32 + mrow;
    const int n1 = n0 + 16;
    const float bias0 = bself[n0], bias1 = bself[n1];

    const char* xbb = (const char*)xb;
    int tile = blockIdx.x;
    // prologue: stage first tile into buf 0
    issue_a_tile(xbb, smemA0, wave, lane, tile * 16 + 2 * wave, tile * 16 + 2 * wave + 1);
    int p = 0;
    for (; tile < NN / 16; tile += gridDim.x) {
        const int nextTile = tile + gridDim.x;
        const bool hasNext = nextTile < NN / 16;
        if (hasNext) {
            issue_a_tile(xbb, smemA0 + (1 - p) * SMEM_A, wave, lane,
                         nextTile * 16 + 2 * wave, nextTile * 16 + 2 * wave + 1);
            wait_async_le2();           // current tile's 2 loads done
        } else {
            wait_async_0();
        }
        __syncthreads();                // tile p visible to all waves

        const char* sA = smemA0 + p * SMEM_A;
        v8f acc0 = {}; v8f acc1 = {};
#pragma unroll
        for (int kk = 0; kk < 8; ++kk) {
            const int kb = kk << 5;
            Frag fa, fb0, fb1;
            const char* ap = sA + mrow * ROWB + ((kb + (half << 4)) << 1);
            fa.q[0] = *(const uint4*)ap;
            fa.q[1] = *(const uint4*)(ap + 16);
            const char* bp0 = smemB + n0 * ROWB + ((kb + (half << 4)) << 1);
            fb0.q[0] = *(const uint4*)bp0;
            fb0.q[1] = *(const uint4*)(bp0 + 16);
            const char* bp1 = smemB + n1 * ROWB + ((kb + (half << 4)) << 1);
            fb1.q[0] = *(const uint4*)bp1;
            fb1.q[1] = *(const uint4*)(bp1 + 16);
            acc0 = __builtin_amdgcn_wmma_f32_16x16x32_bf16(
                false, fa.v, false, fb0.v, (short)0, acc0, false, false);
            acc1 = __builtin_amdgcn_wmma_f32_16x16x32_bf16(
                false, fa.v, false, fb1.v, (short)0, acc1, false, false);
        }

        const int mo = half << 3;
        float* obase = out + (size_t)(tile * 16 + mo) * DD;
#pragma unroll
        for (int i = 0; i < 8; ++i) {
            obase[i * DD + n0] = acc0[i] + bias0;
            obase[i * DD + n1] = acc1[i] + bias1;
        }
        __syncthreads();                // all waves done reading buf p
        p ^= 1;
    }
}

// ---------------------------------------------------------------------------
// edge GEMM: for each relation r (W[r]T resident in LDS), 16-edge tiles:
//   out[dst[e], :] += (bf16(x[src[e], :]) @ bf16(W[r])) / max(deg[r,dst],1)
// ---------------------------------------------------------------------------
__global__ __launch_bounds__(256) void edge_gemm_kernel(
    const unsigned short* __restrict__ xb, const long long* __restrict__ eidx,
    const unsigned short* __restrict__ wt, const float* __restrict__ deg,
    float* __restrict__ out) {
    extern __shared__ char smem[];
    char* smemB  = smem;
    char* smemA0 = smem + SMEM_B;                 // buffer p at smemA0 + p*SMEM_A
    char* metaB  = smem + SMEM_B + 2 * SMEM_A;    // per buffer q: dst at q*128, rdeg at q*128+64

    const int wgPerR = gridDim.x >> 3;       // gridDim.x = 8 * wgPerR
    const int r  = blockIdx.x / wgPerR;
    const int wg = blockIdx.x % wgPerR;

    // stage W[r]T into LDS once
    const unsigned long long* wgp = (const unsigned long long*)(wt + (size_t)r * 65536);
    for (int c = threadIdx.x; c < 16384; c += 256)
        *(unsigned long long*)(smemB + (c >> 6) * ROWB + (c & 63) * 8) = wgp[c];

    const long long* esrc = eidx + (size_t)r * 2 * EE;
    const long long* edst = esrc + EE;
    const float*     degr = deg + (size_t)r * NN;

    const int lane = threadIdx.x & 31;
    const int wave = threadIdx.x >> 5;
    const int mrow = lane & 15;
    const int half = lane >> 4;
    const int n0 = wave * 32 + mrow;
    const int n1 = n0 + 16;
    const char* xbb = (const char*)xb;

    int tile = wg;
    // prologue: stage tile 0 (A rows gathered via src indices) + its meta
    {
        const int eBase = tile * 16;
        int s0 = (int)esrc[eBase + 2 * wave];
        int s1 = (int)esrc[eBase + 2 * wave + 1];
        issue_a_tile(xbb, smemA0, wave, lane, s0, s1);
        if (threadIdx.x < 16) {
            int dd = (int)edst[eBase + threadIdx.x];
            ((int*)metaB)[threadIdx.x] = dd;
            ((float*)(metaB + 64))[threadIdx.x] = 1.0f / fmaxf(degr[dd], 1.0f);
        }
    }
    int p = 0;
    for (; tile < EE / 16; tile += wgPerR) {
        const int nextTile = tile + wgPerR;
        const bool hasNext = nextTile < EE / 16;
        if (hasNext) {
            const int eBase = nextTile * 16;
            int s0 = (int)esrc[eBase + 2 * wave];
            int s1 = (int)esrc[eBase + 2 * wave + 1];
            issue_a_tile(xbb, smemA0 + (1 - p) * SMEM_A, wave, lane, s0, s1);
            if (threadIdx.x < 16) {
                char* mN = metaB + (1 - p) * 128;
                int dd = (int)edst[eBase + threadIdx.x];
                ((int*)mN)[threadIdx.x] = dd;
                ((float*)(mN + 64))[threadIdx.x] = 1.0f / fmaxf(degr[dd], 1.0f);
            }
            wait_async_le2();
        } else {
            wait_async_0();
        }
        __syncthreads();

        const char* sA = smemA0 + p * SMEM_A;
        v8f acc0 = {}; v8f acc1 = {};
#pragma unroll
        for (int kk = 0; kk < 8; ++kk) {
            const int kb = kk << 5;
            Frag fa, fb0, fb1;
            const char* ap = sA + mrow * ROWB + ((kb + (half << 4)) << 1);
            fa.q[0] = *(const uint4*)ap;
            fa.q[1] = *(const uint4*)(ap + 16);
            const char* bp0 = smemB + n0 * ROWB + ((kb + (half << 4)) << 1);
            fb0.q[0] = *(const uint4*)bp0;
            fb0.q[1] = *(const uint4*)(bp0 + 16);
            const char* bp1 = smemB + n1 * ROWB + ((kb + (half << 4)) << 1);
            fb1.q[0] = *(const uint4*)bp1;
            fb1.q[1] = *(const uint4*)(bp1 + 16);
            acc0 = __builtin_amdgcn_wmma_f32_16x16x32_bf16(
                false, fa.v, false, fb0.v, (short)0, acc0, false, false);
            acc1 = __builtin_amdgcn_wmma_f32_16x16x32_bf16(
                false, fa.v, false, fb1.v, (short)0, acc1, false, false);
        }

        // scale by 1/deg and atomically accumulate
        const int*   dstP  = (const int*)(metaB + p * 128);
        const float* rdegP = (const float*)(metaB + p * 128 + 64);
        const int mo = half << 3;
#pragma unroll
        for (int i = 0; i < 8; ++i) {
            const int m = mo + i;
            const int drow = dstP[m];
            const float s = rdegP[m];
            float* pp = out + (size_t)drow * DD;
            atomicAdd(pp + n0, acc0[i] * s);
            atomicAdd(pp + n1, acc1[i] * s);
        }
        __syncthreads();
        p ^= 1;
    }
}

// ---------------------------------------------------------------------------
// finalize: a = silu(h); out = (a-mean)/sqrt(var+eps)*gamma + beta, in place.
// ---------------------------------------------------------------------------
__global__ __launch_bounds__(256) void finalize_kernel(
    float* __restrict__ out, const float* __restrict__ gamma,
    const float* __restrict__ beta) {
    const int node = blockIdx.x * 8 + (threadIdx.x >> 5);
    const int lane = threadIdx.x & 31;
    if (node >= NN) return;
    float* p = out + (size_t)node * DD + lane * 8;
    float4 u0 = *(const float4*)p;
    float4 u1 = *(const float4*)(p + 4);
    float a[8] = {u0.x, u0.y, u0.z, u0.w, u1.x, u1.y, u1.z, u1.w};
    float s = 0.f;
#pragma unroll
    for (int i = 0; i < 8; ++i) {
        a[i] = a[i] / (1.0f + __expf(-a[i]));     // silu
        s += a[i];
    }
#pragma unroll
    for (int o = 16; o > 0; o >>= 1) s += __shfl_xor(s, o, 32);
    const float mean = s * (1.0f / 256.0f);
    float v = 0.f;
#pragma unroll
    for (int i = 0; i < 8; ++i) { a[i] -= mean; v += a[i] * a[i]; }
#pragma unroll
    for (int o = 16; o > 0; o >>= 1) v += __shfl_xor(v, o, 32);
    const float inv = rsqrtf(v * (1.0f / 256.0f) + LN_EPS);
    const float* g = gamma + lane * 8;
    const float* b = beta  + lane * 8;
    float4 w0, w1;
    w0.x = a[0] * inv * g[0] + b[0]; w0.y = a[1] * inv * g[1] + b[1];
    w0.z = a[2] * inv * g[2] + b[2]; w0.w = a[3] * inv * g[3] + b[3];
    w1.x = a[4] * inv * g[4] + b[4]; w1.y = a[5] * inv * g[5] + b[5];
    w1.z = a[6] * inv * g[6] + b[6]; w1.w = a[7] * inv * g[7] + b[7];
    *(float4*)p = w0;
    *(float4*)(p + 4) = w1;
}

// ---------------------------------------------------------------------------
extern "C" void kernel_launch(void* const* d_in, const int* in_sizes, int n_in,
                              void* d_out, int out_size, void* d_ws, size_t ws_size,
                              hipStream_t stream) {
    const float*     x     = (const float*)d_in[0];
    const long long* eidx  = (const long long*)d_in[1];     // (R, 2, E) int64
    const float*     bases = (const float*)d_in[2];
    const float*     coef  = (const float*)d_in[3];
    const float*     wself = (const float*)d_in[4];
    const float*     bself = (const float*)d_in[5];
    const float*     gamma = (const float*)d_in[6];
    const float*     beta  = (const float*)d_in[7];
    float*           out   = (float*)d_out;

    // workspace layout
    char* ws = (char*)d_ws;
    unsigned short* xb  = (unsigned short*)ws;                         // N*256 bf16 = 51.2 MB
    size_t off = (size_t)NN * DD * 2;
    unsigned short* wt  = (unsigned short*)(ws + off);                 // R*65536 bf16 = 1 MB
    off += (size_t)RR * 65536 * 2;
    unsigned short* wst = (unsigned short*)(ws + off);                 // 65536 bf16 = 128 KB
    off += 65536 * 2;
    float*          deg = (float*)(ws + off);                          // R*N f32 = 3.2 MB

    // 1) x -> bf16 once (halves gather traffic, removes per-tile cvt)
    xcvt_kernel<<<NN * DD / 8 / 256, 256, 0, stream>>>(x, xb);

    // 2) transposed bf16 weight tables
    prep_kernel<<<(RR * 65536 + 65536) / 256, 256, 0, stream>>>(bases, coef, wself, wt, wst);

    // 3) degree counts
    zero_kernel<<<1024, 256, 0, stream>>>(deg, RR * NN);
    deg_kernel<<<dim3((EE + 255) / 256, RR), 256, 0, stream>>>(eidx, deg);

    // 4) self GEMM writes h into d_out (WMMA bf16, async double-buffered A)
    self_gemm_kernel<<<1250, 256, SMEM_SELF, stream>>>(xb, wst, bself, out);

    // 5) edge GEMM accumulates normalized messages into d_out
    edge_gemm_kernel<<<RR * 125, 256, SMEM_EDGE, stream>>>(xb, eidx, wt, deg, out);

    // 6) SiLU + LayerNorm in place
    finalize_kernel<<<NN / 8, 256, 0, stream>>>(out, gamma, beta);
}
